// EquiEncoder_40157944217973
// MI455X (gfx1250) — matrix operand order repaired and checked by reading, not compiled, not run on hardware
//
#include <hip/hip_runtime.h>
#include <math.h>

// ---------------- model constants (from reference) ----------------
#define N_ATOMS   12000
#define N_CGB     1500
#define N_NBR     64000
#define N_EDGES   (2 * N_NBR)
#define FEAT      128
#define HALF_FEAT 64
#define N_RBF     16
#define N_CONV    2
#define CUTOFF_F    5.0f
#define CG_CUTOFF_F 12.5f
#define PI_F 3.14159265358979323846f

typedef _Float16 half_t;
typedef __attribute__((ext_vector_type(8)))  _Float16 v8h;
typedef __attribute__((ext_vector_type(16))) _Float16 v16h;
typedef __attribute__((ext_vector_type(8)))  float    v8f;

__device__ __forceinline__ float silu_f(float x) {
    return x / (1.0f + __expf(-x));
}

// ---------------- generic helpers ----------------
__global__ void zero_f(float* __restrict__ p, int n) {
    int i = blockIdx.x * blockDim.x + threadIdx.x;
    if (i < n) p[i] = 0.0f;
}

__global__ void cvt_f16(const float* __restrict__ src, half_t* __restrict__ dst, int n) {
    int i = blockIdx.x * blockDim.x + threadIdx.x;
    if (i < n) dst[i] = (half_t)src[i];
}

// Repack a 128x128 block (cols 0..127) of a row-major f32 weight matrix into
// TRANSPOSED f16: dst[n*128 + k] = src[k*src_stride + n].  Column-major B lets
// each GEMM lane load its 16 B-fragment halves as one contiguous 32B vector.
__global__ void pack_w16_t(const float* __restrict__ src, half_t* __restrict__ dst, int src_stride) {
    int i = blockIdx.x * blockDim.x + threadIdx.x;
    if (i >= FEAT * FEAT) return;
    int k = i >> 7, n = i & (FEAT - 1);
    dst[n * FEAT + k] = (half_t)src[k * src_stride + n];
}

// ---------------- h0 = concat(atom_emb[z], res_emb[cg_z[mapping]]) ----------------
__global__ void init_h(const int* __restrict__ z, const int* __restrict__ cg_z,
                       const int* __restrict__ mapping,
                       const float* __restrict__ atom_emb, const float* __restrict__ res_emb,
                       float* __restrict__ h) {
    int i = blockIdx.x * blockDim.x + threadIdx.x;
    if (i >= N_ATOMS * FEAT) return;
    int a = i >> 7, f = i & (FEAT - 1);
    float val;
    if (f < HALF_FEAT) val = atom_emb[z[a] * HALF_FEAT + f];
    else               val = res_emb[cg_z[mapping[a]] * HALF_FEAT + (f - HALF_FEAT)];
    h[i] = val;
}

__global__ void count_k(const int* __restrict__ mapping, float* __restrict__ cnt) {
    int a = blockIdx.x * blockDim.x + threadIdx.x;
    if (a < N_ATOMS) atomicAdd(&cnt[mapping[a]], 1.0f);
}

// ---------------- WMMA GEMM: out(M x 128) = A(M x 128) @ B(128 x 128) + bias ----------------
// Bt is B TRANSPOSED (Bt[n*128+k]).  SILU_OUT=1: out is half_t with silu applied;
// SILU_OUT=0: out is float.  One wave computes one 16x16 tile; 8 waves/block
// cover all 128 columns; gridDim.x covers the M/16 row tiles.
template<int SILU_OUT>
__global__ __launch_bounds__(256) void gemm128_wmma(const half_t* __restrict__ A,
                                                    const half_t* __restrict__ Bt,
                                                    const float* __restrict__ bias,
                                                    void* __restrict__ out, int M) {
    const int wave = threadIdx.x >> 5;
    const int lane = threadIdx.x & 31;
    const int hi   = lane >> 4;      // half-wave: 0 or 1
    const int mr   = lane & 15;
    const int row0 = blockIdx.x * 16;
    const int col0 = wave * 16;
    if (row0 >= M) return;

    const half_t* arow = A  + (size_t)(row0 + mr) * FEAT;
    const half_t* bcol = Bt + (size_t)(col0 + mr) * FEAT;

    v8f acc = {};
    #pragma unroll
    for (int kc = 0; kc < FEAT / 32; ++kc) {
        const int kbase = kc * 32;
        // A fragment: lane's 16 halves are contiguous at k = kbase+hi*16 .. +15,
        // matching the B fragment's k order.
        v8h alo = *(const v8h*)(arow + kbase + hi * 16);
        v8h ahi = *(const v8h*)(arow + kbase + hi * 16 + 8);
        v16h a, b;
        #pragma unroll
        for (int i = 0; i < 8; ++i) { a[i] = alo[i]; a[i + 8] = ahi[i]; }
        // B fragment: b[i] = B[kbase + hi*16 + i][col0+mr] = Bt[(col0+mr)*128 + kbase + hi*16 + i]
        // -> 16 contiguous halves, one aligned 32B vector load.
        b = *(const v16h*)(bcol + kbase + hi * 16);
        acc = __builtin_amdgcn_wmma_f32_16x16x32_f16(false, a, false, b,
                                                     (short)0, acc, false, false);
    }
    // C/D layout: VGPR r -> (M = hi*8 + r, N = mr)
    #pragma unroll
    for (int r = 0; r < 8; ++r) {
        int m = row0 + hi * 8 + r;
        int n = col0 + mr;
        float val = acc[r] + bias[n];
        if (SILU_OUT) ((half_t*)out)[m * FEAT + n] = (half_t)silu_f(val);
        else          ((float*)out)[m * FEAT + n]  = val;
    }
}

// ---------------- edge message: h[i] += phi[j] * ((rbf(d_ij)@Wd + bd) * env) ----------------
// one block (128 threads) per directed edge
__global__ __launch_bounds__(128) void edge_kernel(const float* __restrict__ xyz,
                                                   const int* __restrict__ nbr,
                                                   const float* __restrict__ phi,
                                                   const float* __restrict__ Wd,   // (16,384)
                                                   const float* __restrict__ bd,   // (384,)
                                                   float* __restrict__ h) {
    __shared__ float s_rbf[N_RBF];
    __shared__ float s_env;
    __shared__ int   s_i, s_j;
    const int e = blockIdx.x;
    const int f = threadIdx.x;
    if (f == 0) {
        int i, j;
        if (e < N_NBR) { i = nbr[2 * e];               j = nbr[2 * e + 1]; }
        else           { int e2 = e - N_NBR; i = nbr[2 * e2 + 1]; j = nbr[2 * e2]; }
        s_i = i; s_j = j;
        float dx = xyz[3 * j + 0] - xyz[3 * i + 0];
        float dy = xyz[3 * j + 1] - xyz[3 * i + 1];
        float dz = xyz[3 * j + 2] - xyz[3 * i + 2];
        float d  = sqrtf(dx * dx + dy * dy + dz * dz);
        s_env = (d < CUTOFF_F) ? 0.5f * (cosf(PI_F * d / CUTOFF_F) + 1.0f) : 0.0f;
        float denom = (d == 0.0f) ? 1.0f : d;
        #pragma unroll
        for (int k = 0; k < N_RBF; ++k)
            s_rbf[k] = sinf((float)(k + 1) * PI_F / CUTOFF_F * d) / denom;
    }
    __syncthreads();
    float acc = bd[f];
    #pragma unroll
    for (int k = 0; k < N_RBF; ++k) acc += s_rbf[k] * Wd[k * 384 + f];
    float s0 = phi[s_j * FEAT + f] * (acc * s_env);
    atomicAdd(&h[s_i * FEAT + f], s0);
}

// ---------------- CG message: Hacc[mapping[a]] += phic[a] * w_c(a) ----------------
__global__ __launch_bounds__(128) void cg_kernel(const float* __restrict__ xyz,
                                                 const float* __restrict__ cg_xyz,
                                                 const int* __restrict__ mapping,
                                                 const float* __restrict__ phic,
                                                 const float* __restrict__ Wd,  // (16,384)
                                                 const float* __restrict__ bd,  // (384,)
                                                 float* __restrict__ Hacc) {
    __shared__ float s_rbf[N_RBF];
    __shared__ float s_env;
    __shared__ int   s_g;
    const int a = blockIdx.x;
    const int f = threadIdx.x;
    if (f == 0) {
        int g = mapping[a];
        s_g = g;
        float dx = xyz[3 * a + 0] - cg_xyz[3 * g + 0];
        float dy = xyz[3 * a + 1] - cg_xyz[3 * g + 1];
        float dz = xyz[3 * a + 2] - cg_xyz[3 * g + 2];
        float d  = sqrtf(dx * dx + dy * dy + dz * dz);
        s_env = (d < CG_CUTOFF_F) ? 0.5f * (cosf(PI_F * d / CG_CUTOFF_F) + 1.0f) : 0.0f;
        float denom = (d == 0.0f) ? 1.0f : d;
        #pragma unroll
        for (int k = 0; k < N_RBF; ++k)
            s_rbf[k] = sinf((float)(k + 1) * PI_F / CG_CUTOFF_F * d) / denom;
    }
    __syncthreads();
    float acc = bd[f];
    #pragma unroll
    for (int k = 0; k < N_RBF; ++k) acc += s_rbf[k] * Wd[k * 384 + f];
    float c0 = phic[a * FEAT + f] * (acc * s_env);
    atomicAdd(&Hacc[s_g * FEAT + f], c0);
}

// ---------------- scatter-mean helpers ----------------
__global__ void scatter_add_k(const float* __restrict__ src, const int* __restrict__ mapping,
                              float* __restrict__ acc) {
    int i = blockIdx.x * blockDim.x + threadIdx.x;
    if (i >= N_ATOMS * FEAT) return;
    int a = i >> 7, f = i & (FEAT - 1);
    atomicAdd(&acc[mapping[a] * FEAT + f], src[i]);
}

__global__ void finalize_k(float* __restrict__ H, const float* __restrict__ acc,
                           const float* __restrict__ cnt, int add) {
    int i = blockIdx.x * blockDim.x + threadIdx.x;
    if (i >= N_CGB * FEAT) return;
    int g = i >> 7;
    float c = fmaxf(cnt[g], 1.0f);
    float v = acc[i] / c;
    H[i] = add ? (H[i] + v) : v;
}

// ---------------- driver ----------------
extern "C" void kernel_launch(void* const* d_in, const int* in_sizes, int n_in,
                              void* d_out, int out_size, void* d_ws, size_t ws_size,
                              hipStream_t stream) {
    (void)in_sizes; (void)n_in; (void)out_size; (void)ws_size;

    const int*   z        = (const int*)d_in[0];
    const float* xyz      = (const float*)d_in[1];
    const int*   cg_z     = (const int*)d_in[2];
    const float* cg_xyz   = (const float*)d_in[3];
    const int*   mapping  = (const int*)d_in[4];
    const int*   nbr      = (const int*)d_in[5];
    /* d_in[6] = cg_nbr_list: unused by the reference */
    const float* atom_emb = (const float*)d_in[7];
    const float* res_emb  = (const float*)d_in[8];
    const float* msg_W1   = (const float*)d_in[9];
    const float* msg_b1   = (const float*)d_in[10];
    const float* msg_W2   = (const float*)d_in[11];
    const float* msg_b2   = (const float*)d_in[12];
    const float* msg_Wd   = (const float*)d_in[13];
    const float* msg_bd   = (const float*)d_in[14];
    const float* cg_W1    = (const float*)d_in[15];
    const float* cg_b1    = (const float*)d_in[16];
    const float* cg_W2    = (const float*)d_in[17];
    const float* cg_b2    = (const float*)d_in[18];
    const float* cg_Wd    = (const float*)d_in[19];
    const float* cg_bd    = (const float*)d_in[20];

    float* H = (float*)d_out;              // (1500, 128)
    float* h = H + N_CGB * FEAT;           // (12000, 128)

    // workspace carve-out (256B aligned)
    char* wp = (char*)d_ws;
    auto carve = [&](size_t bytes) -> void* {
        void* r = (void*)wp;
        wp += (bytes + 255) & ~(size_t)255;
        return r;
    };
    half_t* h16    = (half_t*)carve((size_t)N_ATOMS * FEAT * sizeof(half_t));
    half_t* hid16  = (half_t*)carve((size_t)N_ATOMS * FEAT * sizeof(half_t));
    float*  phi    = (float*)carve((size_t)N_ATOMS * FEAT * sizeof(float));
    float*  Hacc   = (float*)carve((size_t)N_CGB * FEAT * sizeof(float));
    float*  cnt    = (float*)carve((size_t)N_CGB * sizeof(float));
    half_t* W1f    = (half_t*)carve((size_t)N_CONV * FEAT * FEAT * sizeof(half_t));
    half_t* W2f    = (half_t*)carve((size_t)N_CONV * FEAT * FEAT * sizeof(half_t));
    half_t* cW1f   = (half_t*)carve((size_t)N_CONV * FEAT * FEAT * sizeof(half_t));
    half_t* cW2f   = (half_t*)carve((size_t)N_CONV * FEAT * FEAT * sizeof(half_t));

    const int T = 256;
    const int nHF  = N_CGB * FEAT;     // 192000
    const int nAF  = N_ATOMS * FEAT;   // 1536000
    const int gHF  = (nHF + T - 1) / T;
    const int gAF  = (nAF + T - 1) / T;
    const int gWW  = (FEAT * FEAT + T - 1) / T;
    const int gGEMM = N_ATOMS / 16;    // 750 row tiles

    // init: H=0, cnt, h0
    zero_f<<<gHF, T, 0, stream>>>(H, nHF);
    zero_f<<<(N_CGB + T - 1) / T, T, 0, stream>>>(cnt, N_CGB);
    init_h<<<gAF, T, 0, stream>>>(z, cg_z, mapping, atom_emb, res_emb, h);
    count_k<<<(N_ATOMS + T - 1) / T, T, 0, stream>>>(mapping, cnt);

    // pack weights to transposed f16 (only cols 0..127 of W2/cg_W2 are live)
    for (int l = 0; l < N_CONV; ++l) {
        pack_w16_t<<<gWW, T, 0, stream>>>(msg_W1 + (size_t)l * FEAT * FEAT, W1f + (size_t)l * FEAT * FEAT, FEAT);
        pack_w16_t<<<gWW, T, 0, stream>>>(msg_W2 + (size_t)l * FEAT * 384,  W2f + (size_t)l * FEAT * FEAT, 384);
        pack_w16_t<<<gWW, T, 0, stream>>>(cg_W1  + (size_t)l * FEAT * FEAT, cW1f + (size_t)l * FEAT * FEAT, FEAT);
        pack_w16_t<<<gWW, T, 0, stream>>>(cg_W2  + (size_t)l * FEAT * 384,  cW2f + (size_t)l * FEAT * FEAT, 384);
    }

    for (int l = 0; l < N_CONV; ++l) {
        // phi = silu(h@W1+b1) @ W2[:, :128] + b2[:128]
        cvt_f16<<<gAF, T, 0, stream>>>(h, h16, nAF);
        gemm128_wmma<1><<<gGEMM, 256, 0, stream>>>(h16, W1f + (size_t)l * FEAT * FEAT,
                                                   msg_b1 + (size_t)l * FEAT, hid16, N_ATOMS);
        gemm128_wmma<0><<<gGEMM, 256, 0, stream>>>(hid16, W2f + (size_t)l * FEAT * FEAT,
                                                   msg_b2 + (size_t)l * 384, phi, N_ATOMS);
        // h[i] += sum over edges of phi[j] * w_s
        edge_kernel<<<N_EDGES, 128, 0, stream>>>(xyz, nbr, phi,
                                                 msg_Wd + (size_t)l * N_RBF * 384,
                                                 msg_bd + (size_t)l * 384, h);
        if (l == 0) {
            // H = scatter_mean(h, mapping)
            zero_f<<<gHF, T, 0, stream>>>(Hacc, nHF);
            scatter_add_k<<<gAF, T, 0, stream>>>(h, mapping, Hacc);
            finalize_k<<<gHF, T, 0, stream>>>(H, Hacc, cnt, /*add=*/0);
        }
        // phic = silu(h@cgW1+cgb1) @ cgW2[:, :128] + cgb2[:128]
        cvt_f16<<<gAF, T, 0, stream>>>(h, h16, nAF);
        gemm128_wmma<1><<<gGEMM, 256, 0, stream>>>(h16, cW1f + (size_t)l * FEAT * FEAT,
                                                   cg_b1 + (size_t)l * FEAT, hid16, N_ATOMS);
        gemm128_wmma<0><<<gGEMM, 256, 0, stream>>>(hid16, cW2f + (size_t)l * FEAT * FEAT,
                                                   cg_b2 + (size_t)l * 384, phi, N_ATOMS);
        // H += scatter_mean(c0, mapping)
        zero_f<<<gHF, T, 0, stream>>>(Hacc, nHF);
        cg_kernel<<<N_ATOMS, 128, 0, stream>>>(xyz, cg_xyz, mapping, phi,
                                               cg_Wd + (size_t)l * N_RBF * 384,
                                               cg_bd + (size_t)l * 384, Hacc);
        finalize_k<<<gHF, T, 0, stream>>>(H, Hacc, cnt, /*add=*/1);
    }
}
